// TriangleAttention_9775345565946
// MI455X (gfx1250) — matrix operand order repaired and checked by reading, not compiled, not run on hardware
//
#include <hip/hip_runtime.h>
#include <hip/hip_bf16.h>

typedef __attribute__((ext_vector_type(16))) __bf16 v16bf;
typedef __attribute__((ext_vector_type(8)))  float  v8f;
typedef __attribute__((ext_vector_type(4)))  unsigned int v4u;
typedef __attribute__((ext_vector_type(8)))  int  v8i;
typedef __attribute__((ext_vector_type(4)))  int  v4i;

#define N_SEQ 256
#define C_DIM 128
#define H_HEADS 4
#define D_HEAD 32

#if __has_builtin(__builtin_amdgcn_tensor_load_to_lds) && \
    __has_builtin(__builtin_amdgcn_s_wait_tensorcnt)
#define USE_TDM 1
#endif

__device__ __forceinline__ v8f wmma_bf16(v16bf a, v16bf b, v8f c) {
  // (neg_a, A, neg_b, B, c_mod, C, reuse_a, reuse_b)
  return __builtin_amdgcn_wmma_f32_16x16x32_bf16(false, a, false, b, (short)0, c,
                                                 false, false);
}

union FragBF {
  v16bf v;
  __bf16 e[16];
  uint4 q[2];
};

// Low 32 bits of a generic pointer to __shared__ data == LDS byte offset
// (LDS aperture maps addr[31:0] directly; see CDNA5 ISA).
__device__ __forceinline__ unsigned lds_addr_of(const void* p) {
  return (unsigned)(unsigned long long)p;
}

#ifdef USE_TDM
// Issue a TDM 1-D tile copy: n_elems 16-bit elements, global -> LDS.
// D# built per CDNA5 ISA (group0) / §8.4 (group1).
// This toolchain's builtin takes 6 args:
//   (v4u group0, v8i group1, v4i group2, v4i group3, v8i extra, i32 cpol)
__device__ __forceinline__ void tdm_load_u16(unsigned lds_byte_addr,
                                             const void* gsrc, unsigned n_elems) {
  unsigned long long ga = (unsigned long long)gsrc;
  v4u g0;
  g0.x = 1u;                                              // count=1, user descriptor
  g0.y = lds_byte_addr;                                   // lds_addr
  g0.z = (unsigned)(ga & 0xffffffffu);                    // global_addr[31:0]
  g0.w = (unsigned)((ga >> 32) & 0x01ffffffu) | (2u << 30);  // ga[56:32] | type=2
  v8i g1;
  g1[0] = (int)(1u << 16);                  // workgroup_mask=0, data_size=1 (2B)
  g1[1] = (int)((n_elems & 0xffffu) << 16); // tensor_dim0[15:0] @ bits 63:48
  g1[2] = (int)(((n_elems >> 16) & 0xffffu) | (1u << 16)); // dim0 hi | tensor_dim1=1
  g1[3] = (int)((n_elems & 0xffffu) << 16); // tile_dim0 @ bits 127:112
  g1[4] = 1;                                // tile_dim1=1, tile_dim2=0
  g1[5] = (int)n_elems;                     // tensor_dim0_stride[31:0]
  g1[6] = 0;                                // stride0 hi, tensor_dim1_stride lo
  g1[7] = 0;
  v4i z4 = {0, 0, 0, 0};
  v8i z8 = {0, 0, 0, 0, 0, 0, 0, 0};
  __builtin_amdgcn_tensor_load_to_lds(g0, g1, z4, z4, z8, 0);
}
#endif

// A fragment (16x32, M x K) from fp32 row-major [.. x lda], converted to bf16.
// ISA layout: lanes 0-15 row m=lane, K {kc..kc+7, kc+16..kc+23};
//             lanes 16-31 row m=lane-16, K {kc+8..kc+15, kc+24..kc+31}.
__device__ __forceinline__ v16bf load_a_f32(const float* A, int lda, int row_base,
                                            int kc, int lane) {
  int half = (lane >> 4) & 1;
  int m = lane & 15;
  const float* rp = A + (size_t)(row_base + m) * lda;
  int k0 = kc + 8 * half;
  int k1 = kc + 16 + 8 * half;
  const float4* p0 = (const float4*)(rp + k0);
  const float4* p1 = (const float4*)(rp + k1);
  float4 f0 = p0[0], f1 = p0[1], f2 = p1[0], f3 = p1[1];
  FragBF u;
  u.e[0] = (__bf16)f0.x; u.e[1] = (__bf16)f0.y; u.e[2] = (__bf16)f0.z; u.e[3] = (__bf16)f0.w;
  u.e[4] = (__bf16)f1.x; u.e[5] = (__bf16)f1.y; u.e[6] = (__bf16)f1.z; u.e[7] = (__bf16)f1.w;
  u.e[8] = (__bf16)f2.x; u.e[9] = (__bf16)f2.y; u.e[10] = (__bf16)f2.z; u.e[11] = (__bf16)f2.w;
  u.e[12] = (__bf16)f3.x; u.e[13] = (__bf16)f3.y; u.e[14] = (__bf16)f3.z; u.e[15] = (__bf16)f3.w;
  return u.v;
}

// A fragment from bf16 row-major [.. x lda] (rows 16B aligned).
__device__ __forceinline__ v16bf load_a_bf16(const __bf16* A, int lda, int row_base,
                                             int kc, int lane) {
  int half = (lane >> 4) & 1;
  int m = lane & 15;
  const __bf16* rp = A + (size_t)(row_base + m) * lda;
  FragBF u;
  u.q[0] = *(const uint4*)(rp + kc + 8 * half);
  u.q[1] = *(const uint4*)(rp + kc + 16 + 8 * half);
  return u.v;
}

// B fragment (32x16, K x N) from fp32 row-major W[K x ldb], converted to bf16.
// ISA layout: lanes 0-15 col n=lane hold K {kc..kc+7, kc+16..kc+23}; lanes 16-31 hold K {kc+8..kc+15, kc+24..kc+31}.
__device__ __forceinline__ v16bf load_b_f32(const float* W, int ldb, int kc,
                                            int col_base, int lane) {
  int half = (lane >> 4) & 1;
  int n = col_base + (lane & 15);
  int kb = kc + 8 * half;
  FragBF u;
#pragma unroll
  for (int t = 0; t < 16; ++t)
    u.e[t] = (__bf16)W[(size_t)(kb + t + (t >= 8 ? 8 : 0)) * ldb + n];
  return u.v;
}

// ---------------- Kernel 1: QKV projection ----------------
// qkv = pair_rep[65536 x 128] @ Wqkv[128 x 384] + bqkv
// Each wave: one 16-row M-tile x 6 N-tiles (A fragments loaded once, 6x reuse).
__global__ __launch_bounds__(256) void qkv_gemm(const float* __restrict__ pair,
                                                const float* __restrict__ Wqkv,
                                                const float* __restrict__ bqkv,
                                                __bf16* __restrict__ qws,
                                                __bf16* __restrict__ kws,
                                                __bf16* __restrict__ vws) {
  int lane = threadIdx.x & 31;
  int wave = threadIdx.x >> 5;
  int job = blockIdx.x * 8 + wave;   // 4096 M-tiles * 4 column-groups
  int cg = job & 3;
  int mt = job >> 2;
  int row_base = mt * 16;
  int half = (lane >> 4) & 1;

  v16bf a[4];
#pragma unroll
  for (int c = 0; c < 4; ++c)
    a[c] = load_a_f32(pair, C_DIM, row_base, c * 32, lane);

  for (int t6 = 0; t6 < 6; ++t6) {
    int nt = cg * 6 + t6;
    int col_base = nt * 16;
    v8f acc = {};
#pragma unroll
    for (int c = 0; c < 4; ++c)
      acc = wmma_bf16(a[c], load_b_f32(Wqkv, 3 * C_DIM, c * 32, col_base, lane), acc);

    int nn = col_base + (lane & 15);
    float bias = bqkv[nn];
    int which = nn / C_DIM;            // 0=q 1=k 2=v (constant per tile)
    int hd = nn % C_DIM;
    int h = hd >> 5;
    int d = hd & 31;
    __bf16* dst = (which == 0) ? qws : (which == 1) ? kws : vws;
#pragma unroll
    for (int r = 0; r < 8; ++r) {
      int row = row_base + r + 8 * half;  // C layout: VGPR r -> M = r (+8 hi half)
      int i = row >> 8;
      int j = row & 255;
      size_t off = ((((size_t)i * H_HEADS + h) * N_SEQ) + j) * D_HEAD + d;
      dst[off] = (__bf16)(acc[r] + bias);
    }
  }
}

// ---------------- Kernel 2: fused attention per (h,i) ----------------
__global__ __launch_bounds__(256) void attn(const __bf16* __restrict__ qws,
                                            const __bf16* __restrict__ kws,
                                            const __bf16* __restrict__ vws,
                                            const float* __restrict__ dm,
                                            const float* __restrict__ dscale,
                                            const unsigned char* __restrict__ mask,
                                            __bf16* __restrict__ ows) {
  __shared__ __bf16 Ks[N_SEQ * D_HEAD];    // 16 KB
  __shared__ __bf16 Vs[N_SEQ * D_HEAD];    // 16 KB
  __shared__ __bf16 Ps[8][16 * 32];        // 8 KB, per-wave P staging

  int bx = blockIdx.x;                     // N_SEQ * H_HEADS blocks
  int h = bx & (H_HEADS - 1);
  int i = bx >> 2;

  const __bf16* Kg = kws + (((size_t)i * H_HEADS + h) * N_SEQ) * D_HEAD;
  const __bf16* Vg = vws + (((size_t)i * H_HEADS + h) * N_SEQ) * D_HEAD;

#ifdef USE_TDM
  // Stage K/V via the Tensor Data Mover: one wave issues two 16 KB tile copies,
  // waits on TENSORcnt, then the block synchronizes.
  if (threadIdx.x < 32) {
    tdm_load_u16(lds_addr_of(Ks), Kg, N_SEQ * D_HEAD);
    tdm_load_u16(lds_addr_of(Vs), Vg, N_SEQ * D_HEAD);
    __builtin_amdgcn_s_wait_tensorcnt(0);
  }
  __syncthreads();
#else
  {
    const uint4* ks4 = (const uint4*)Kg;
    const uint4* vs4 = (const uint4*)Vg;
    uint4* Ks4 = (uint4*)Ks;
    uint4* Vs4 = (uint4*)Vs;
    for (int t = threadIdx.x; t < (N_SEQ * D_HEAD) / 8; t += 256) {
      Ks4[t] = ks4[t];
      Vs4[t] = vs4[t];
    }
  }
  __syncthreads();
#endif

  int lane = threadIdx.x & 31;
  int wave = threadIdx.x >> 5;
  int half = (lane >> 4) & 1;
  int l = lane & 15;
  const float scale = 0.17677669529663687f;  // 1/sqrt(32)
  float ds = dscale[0];

  for (int jt = wave; jt < 16; jt += 8) {
    __builtin_prefetch(dm + (size_t)(jt * 16) * N_SEQ, 0, 0);
    __builtin_prefetch(mask + (size_t)(jt * 16) * N_SEQ, 0, 0);

    // ---- Q A-fragment (16 x 32, full D) directly from global bf16 ----
    FragBF aq;
    {
      int jr = jt * 16 + l;
      const uint4* qp =
          (const uint4*)(qws + ((((size_t)i * H_HEADS + h) * N_SEQ) + jr) * D_HEAD);
      aq.q[0] = qp[half];      // d 0..7 or 8..15
      aq.q[1] = qp[2 + half];  // d 16..23 or 24..31
    }

    // ---- scores: 16 accumulator tiles of 16x16, one WMMA each (K = D = 32) ----
    v8f s[16];
#pragma unroll
    for (int kt = 0; kt < 16; ++kt) {
      FragBF bk;
      const uint4* kp = (const uint4*)(Ks + (kt * 16 + l) * D_HEAD);
      bk.q[0] = kp[half];
      bk.q[1] = kp[2 + half];
      v8f z = {};
      s[kt] = wmma_bf16(aq.v, bk.v, z);
    }

    // ---- scale + distance bias + mask, row max ----
    float rmax[8];
#pragma unroll
    for (int r = 0; r < 8; ++r) rmax[r] = -__builtin_inff();
#pragma unroll
    for (int kt = 0; kt < 16; ++kt) {
#pragma unroll
      for (int r = 0; r < 8; ++r) {
        int j = jt * 16 + r + 8 * half;
        int kk = kt * 16 + l;
        float v = s[kt][r] * scale + ds * dm[j * N_SEQ + kk];
        if (mask[j * N_SEQ + kk]) v = -__builtin_inff();
        s[kt][r] = v;
        rmax[r] = fmaxf(rmax[r], v);
      }
    }
    // reduce within each 16-lane half (rows r and r+8 reduced independently)
#pragma unroll
    for (int r = 0; r < 8; ++r) {
#pragma unroll
      for (int m = 1; m < 16; m <<= 1)
        rmax[r] = fmaxf(rmax[r], __shfl_xor(rmax[r], m, 32));
    }

    // ---- exp + row sum ----
    float rsum[8];
#pragma unroll
    for (int r = 0; r < 8; ++r) rsum[r] = 0.f;
#pragma unroll
    for (int kt = 0; kt < 16; ++kt) {
#pragma unroll
      for (int r = 0; r < 8; ++r) {
        float e = __expf(s[kt][r] - rmax[r]);
        s[kt][r] = e;
        rsum[r] += e;
      }
    }
#pragma unroll
    for (int r = 0; r < 8; ++r) {
#pragma unroll
      for (int m = 1; m < 16; m <<= 1)
        rsum[r] += __shfl_xor(rsum[r], m, 32);
      rsum[r] = 1.f / rsum[r];
    }

    // ---- O = P @ V over 8 k-chunks of 32 ----
    v8f o0 = {}, o1 = {};
    for (int g = 0; g < 8; ++g) {
      // P chunk: C layout -> LDS [m][k_local], reread as A fragment (same wave,
      // DScnt ordering handled by compiler-inserted waits).
#pragma unroll
      for (int t = 0; t < 2; ++t) {
        int kt = 2 * g + t;
#pragma unroll
        for (int r = 0; r < 8; ++r) {
          int m = r + 8 * half;
          Ps[wave][m * 32 + t * 16 + l] = (__bf16)(s[kt][r] * rsum[r]);
        }
      }
      FragBF ap;
      {
        const uint4* pp = (const uint4*)(Ps[wave] + l * 32);
        ap.q[0] = pp[2 * half];
        ap.q[1] = pp[2 * half + 1];
      }
      FragBF bv0, bv1;
#pragma unroll
      for (int t = 0; t < 16; ++t) {
        int krow = g * 32 + 16 * half + t;
        bv0.e[t] = Vs[krow * D_HEAD + l];
        bv1.e[t] = Vs[krow * D_HEAD + 16 + l];
      }
      o0 = wmma_bf16(ap.v, bv0.v, o0);
      o1 = wmma_bf16(ap.v, bv1.v, o1);
    }

    // ---- store O as bf16, layout [i*256+j][h*32+d] ----
#pragma unroll
    for (int r = 0; r < 8; ++r) {
      int j = jt * 16 + r + 8 * half;
      size_t base = (((size_t)i * N_SEQ + j) * C_DIM) + h * D_HEAD;
      ows[base + l] = (__bf16)o0[r];
      ows[base + 16 + l] = (__bf16)o1[r];
    }
  }
}

// ---------------- Kernel 3: output projection ----------------
// out = O[65536 x 128](bf16) @ Wout[128 x 128] + bout -> fp32
// Each wave: one M-tile x all 8 N-tiles (A fragments loaded once, 8x reuse).
__global__ __launch_bounds__(256) void out_gemm(const __bf16* __restrict__ ows,
                                                const float* __restrict__ Wout,
                                                const float* __restrict__ bout,
                                                float* __restrict__ out) {
  int lane = threadIdx.x & 31;
  int wave = threadIdx.x >> 5;
  int mt = blockIdx.x * 8 + wave;    // 4096 M-tiles
  int row_base = mt * 16;
  int half = (lane >> 4) & 1;

  v16bf a[4];
#pragma unroll
  for (int c = 0; c < 4; ++c)
    a[c] = load_a_bf16(ows, C_DIM, row_base, c * 32, lane);

  for (int nt = 0; nt < 8; ++nt) {
    int col_base = nt * 16;
    v8f acc = {};
#pragma unroll
    for (int c = 0; c < 4; ++c)
      acc = wmma_bf16(a[c], load_b_f32(Wout, C_DIM, c * 32, col_base, lane), acc);

    int n = col_base + (lane & 15);
    float bias = bout[n];
#pragma unroll
    for (int r = 0; r < 8; ++r) {
      int row = row_base + r + 8 * half;
      out[(size_t)row * C_DIM + n] = acc[r] + bias;
    }
  }
}

extern "C" void kernel_launch(void* const* d_in, const int* in_sizes, int n_in,
                              void* d_out, int out_size, void* d_ws, size_t ws_size,
                              hipStream_t stream) {
  const float* pair = (const float*)d_in[0];           // [256,256,128]
  const float* dm = (const float*)d_in[1];             // [256,256]
  const float* Wqkv = (const float*)d_in[2];           // [128,384]
  const float* bqkv = (const float*)d_in[3];           // [384]
  const float* Wout = (const float*)d_in[4];           // [128,128]
  const float* bout = (const float*)d_in[5];           // [128]
  const float* dscale = (const float*)d_in[6];         // [1]
  const unsigned char* mask = (const unsigned char*)d_in[7];  // [256,256] bool

  // workspace: q/k/v each [N][H][N][D] bf16 (16 MB), O [N*N][C] bf16 (16 MB)
  const size_t qkv_elems = (size_t)N_SEQ * H_HEADS * N_SEQ * D_HEAD;  // 8388608
  __bf16* qws = (__bf16*)d_ws;
  __bf16* kws = qws + qkv_elems;
  __bf16* vws = kws + qkv_elems;
  __bf16* ows = vws + qkv_elems;

  // 1) QKV projection: 4096 M-tiles x 4 column-groups, 8 wave-jobs/block
  qkv_gemm<<<dim3((4096 * 4) / 8), dim3(256), 0, stream>>>(pair, Wqkv, bqkv, qws,
                                                           kws, vws);
  // 2) fused attention, one block per (h, i)
  attn<<<dim3(N_SEQ * H_HEADS), dim3(256), 0, stream>>>(qws, kws, vws, dm, dscale,
                                                        mask, ows);
  // 3) output projection: 4096 M-tile wave-jobs, 8 per block
  out_gemm<<<dim3(4096 / 8), dim3(256), 0, stream>>>(ows, Wout, bout,
                                                     (float*)d_out);
}
